// VC_DimLoss_40415642255434
// MI455X (gfx1250) — compile-verified, matrix operand order checked by reading, not run on hardware
//
#include <hip/hip_runtime.h>
#include <hip/hip_bf16.h>

// MI455X (gfx1250) implementation.
// One persistent 512-thread workgroup per class pair (45 blocks). alpha/beta
// live in LDS (2 x 40KB of the 320KB/WGP). X (100MB) is L2-resident (192MB
// L2). Gradient matvecs use v_wmma_f32_16x16x32_f16 (f16 in, f32 accumulate).
// Simplex projection via bisection on the water-filling threshold (identical
// result to the sort-based reference formula).
//
// This revision forces address spaces explicitly: LDS ptrs are AS(3) so all
// on-chip state uses ds_load/ds_store, and X/idx are AS(1) so matrix rows use
// global_load_b128 -- no flat loads, no fused loadcnt+dscnt waits. The WMMA
// B operand (diff in column N=0) is built branch-free into registers once per
// PGD iteration and reused by both gradient passes.

#define NUM_CLASSES 10
#define D 256
#define MAX_ITE 100
#define MIN_GAP 1e-10f
#define DECAY_EVERY 10
#define DECAY 0.9f
#define LR0 0.01f
#define BLOCK 512
#define NWAVES (BLOCK / 32)
#define BISECT_ITERS 40

typedef __attribute__((ext_vector_type(16))) _Float16 v16h;
typedef __attribute__((ext_vector_type(8)))  float    v8f;
typedef __attribute__((ext_vector_type(4)))  float    f32x4;

// explicit address-space pointer element types
typedef __attribute__((address_space(3))) float        lfloat;   // LDS
typedef __attribute__((address_space(1))) const float  gcfloat;  // global ro
typedef __attribute__((address_space(1))) const int    gcint;    // global ro
typedef __attribute__((address_space(1))) const f32x4  gcf32x4;  // global ro vec

// ---------------- fixed-order reductions (deterministic) ----------------

__device__ __forceinline__ float waveRedSum(float v) {
#pragma unroll
  for (int o = 16; o > 0; o >>= 1) v += __shfl_down(v, o, 32);
  return v;
}
__device__ __forceinline__ float waveRedMax(float v) {
#pragma unroll
  for (int o = 16; o > 0; o >>= 1) v = fmaxf(v, __shfl_down(v, o, 32));
  return v;
}
__device__ __forceinline__ float waveRedMin(float v) {
#pragma unroll
  for (int o = 16; o > 0; o >>= 1) v = fminf(v, __shfl_down(v, o, 32));
  return v;
}

__device__ __forceinline__ float blockSum(float v, lfloat* s_red) {
  int lane = threadIdx.x & 31, wid = threadIdx.x >> 5;
  v = waveRedSum(v);
  if (lane == 0) s_red[wid] = v;
  __syncthreads();
  float r = (threadIdx.x < NWAVES) ? s_red[threadIdx.x] : 0.0f;
  if (wid == 0) { r = waveRedSum(r); if (lane == 0) s_red[0] = r; }
  __syncthreads();
  r = s_red[0];
  __syncthreads();
  return r;
}

__device__ __forceinline__ float blockMax(float v, lfloat* s_red) {
  int lane = threadIdx.x & 31, wid = threadIdx.x >> 5;
  v = waveRedMax(v);
  if (lane == 0) s_red[wid] = v;
  __syncthreads();
  float r = (threadIdx.x < NWAVES) ? s_red[threadIdx.x] : -3.4e38f;
  if (wid == 0) { r = waveRedMax(r); if (lane == 0) s_red[0] = r; }
  __syncthreads();
  r = s_red[0];
  __syncthreads();
  return r;
}

__device__ __forceinline__ float blockMin(float v, lfloat* s_red) {
  int lane = threadIdx.x & 31, wid = threadIdx.x >> 5;
  v = waveRedMin(v);
  if (lane == 0) s_red[wid] = v;
  __syncthreads();
  float r = (threadIdx.x < NWAVES) ? s_red[threadIdx.x] : 3.4e38f;
  if (wid == 0) { r = waveRedMin(r); if (lane == 0) s_red[0] = r; }
  __syncthreads();
  r = s_red[0];
  __syncthreads();
  return r;
}

// ---------------- diff = A^T alpha - B^T beta (d=256) ----------------
// 512 threads = 2 row-groups x 256 columns; coalesced 1KB row reads (L2 hits).

__device__ __forceinline__ void compute_diff(gcfloat* X,
                                             gcint* idxA, gcint* idxB,
                                             lfloat* s_alpha, lfloat* s_beta,
                                             lfloat* s_diff, lfloat* s_red,
                                             int m) {
  int tid = threadIdx.x;
  int col = tid & (D - 1);
  int grp = tid >> 8;               // 0..1
  float acc = 0.0f;
#pragma unroll 4
  for (int k = grp; k < m; k += (BLOCK / D)) {
    gcfloat* ra = X + (size_t)idxA[k] * D;
    gcfloat* rb = X + (size_t)idxB[k] * D;
    acc = fmaf(s_alpha[k], ra[col], acc);
    acc = fmaf(-s_beta[k], rb[col], acc);
  }
  s_red[tid] = acc;
  __syncthreads();
  if (tid < D) s_diff[tid] = s_red[tid] + s_red[tid + D];
  __syncthreads();
}

// ---------------- WMMA B operand: diff chunk in column N=0 ----------------
// ISA 16-bit operand layout: lane.N = lane&15, K-half h = lane>>4,
// VGPR v holds K pair kb(v,h) = ((v<4)?2v:2v+8)+8h. Built branch-free with a
// multiplicative lane mask (only lanes 0 and 16 carry column 0).

__device__ __forceinline__ void build_bmat(lfloat* s_diff, v16h* bmat) {
  int lane = threadIdx.x & 31;
  int Mh = lane & 15, h = lane >> 4;
  float msk = (Mh == 0) ? 1.0f : 0.0f;
#pragma unroll
  for (int ch = 0; ch < 8; ++ch) {
    int kbase = ch << 5;
#pragma unroll
    for (int v = 0; v < 8; ++v) {
      int kb = ((v < 4) ? (2 * v) : (2 * v + 8)) + 8 * h;
      bmat[ch][2 * v]     = (_Float16)(msk * s_diff[kbase + kb]);
      bmat[ch][2 * v + 1] = (_Float16)(msk * s_diff[kbase + kb + 1]);
    }
  }
}

// ---------------- gradient via WMMA: s_vec[k] -= coef * dot(X[idx[k]], diff) --
// 16-row tiles per wave; per lane, the K pattern per 32-chunk is two
// contiguous 8-float runs -> four global_load_b128 per chunk.

__device__ __forceinline__ void grad_update(gcfloat* X, gcint* idxC,
                                            lfloat* s_vec, const v16h* bmat,
                                            float coef, int m) {
  int tid  = threadIdx.x;
  int lane = tid & 31;
  int wid  = tid >> 5;
  int Mh   = lane & 15;             // M coordinate (rows of the tile)
  int h    = lane >> 4;             // K-half selector
  int ntiles = m >> 4;              // 625 tiles of 16 rows

  for (int t = wid; t < ntiles; t += NWAVES) {
    int rowbase = t << 4;
    gcfloat* rp = X + (size_t)idxC[rowbase + Mh] * D;
    v8f acc = {};
#pragma unroll
    for (int ch = 0; ch < 8; ++ch) {
      int kbase = ch << 5;
      f32x4 q0 = *(gcf32x4*)(rp + kbase + 8 * h);
      f32x4 q1 = *(gcf32x4*)(rp + kbase + 8 * h + 4);
      f32x4 q2 = *(gcf32x4*)(rp + kbase + 16 + 8 * h);
      f32x4 q3 = *(gcf32x4*)(rp + kbase + 16 + 8 * h + 4);
      v16h a;
#pragma unroll
      for (int j = 0; j < 4; ++j) {
        a[j]      = (_Float16)q0[j];
        a[4 + j]  = (_Float16)q1[j];
        a[8 + j]  = (_Float16)q2[j];
        a[12 + j] = (_Float16)q3[j];
      }
      acc = __builtin_amdgcn_wmma_f32_16x16x32_f16(
          false, a, false, bmat[ch], (short)0, acc, false, false);
    }
    if (Mh == 0) {                  // lane 0: rows +0..7, lane 16: rows +8..15
#pragma unroll
      for (int r = 0; r < 8; ++r) {
        int rr = rowbase + r + 8 * h;
        s_vec[rr] -= coef * acc[r];
      }
    }
  }
  __syncthreads();
}

// ---------------- simplex projection via threshold bisection ----------------
// Finds tau with sum(max(v - tau, 0)) == 1; identical to sort-based formula.

__device__ __forceinline__ void project_simplex_lds(lfloat* v, lfloat* s_red,
                                                    int m) {
  int tid = threadIdx.x;
  float mx = -3.4e38f, mn = 3.4e38f;
  for (int k = tid; k < m; k += BLOCK) {
    float x = v[k];
    mx = fmaxf(mx, x);
    mn = fminf(mn, x);
  }
  mx = blockMax(mx, s_red);
  mn = blockMin(mn, s_red);
  float lo = mn - 1.0f;   // sum(lo) >= m >= 1
  float hi = mx;          // sum(hi) == 0 < 1
#pragma unroll 1
  for (int it = 0; it < BISECT_ITERS; ++it) {
    float mid = 0.5f * (lo + hi);
    float s = 0.0f;
    for (int k = tid; k < m; k += BLOCK) s += fmaxf(v[k] - mid, 0.0f);
    s = blockSum(s, s_red);
    if (s > 1.0f) lo = mid; else hi = mid;   // uniform branch
  }
  float tau = 0.5f * (lo + hi);
  for (int k = tid; k < m; k += BLOCK) v[k] = fmaxf(v[k] - tau, 0.0f);
  __syncthreads();
}

// ---------------- stable per-class index lists (mirrors stable argsort) -----

__global__ void build_idx_kernel(const int* __restrict__ y, int* __restrict__ idx,
                                 int n, int m) {
  int c = threadIdx.x;
  if (c >= NUM_CLASSES) return;
  int cnt = 0;
  for (int i = 0; i < n; ++i) {
    if (y[i] == c && cnt < m) idx[c * m + cnt++] = i;
  }
}

// ---------------- per-pair PGD solver ----------------

__global__ void __launch_bounds__(BLOCK)
vc_pair_kernel(const float* __restrict__ Xp, const int* __restrict__ idxp,
               float* __restrict__ dis, int m) {
  extern __shared__ float smem_raw[];
  lfloat* smem = (lfloat*)smem_raw;
  lfloat* s_alpha = smem;                 // m
  lfloat* s_beta  = smem + m;             // m
  lfloat* s_diff  = smem + 2 * m;         // D
  lfloat* s_red   = smem + 2 * m + D;     // BLOCK

  gcfloat* X  = (gcfloat*)Xp;
  gcint* idx  = (gcint*)idxp;

  // triu_indices(k=1) pair decode
  int p = blockIdx.x;
  int ci = 0, rem = p, span = NUM_CLASSES - 1;
  while (rem >= span) { rem -= span; ++ci; --span; }
  int cj = ci + 1 + rem;
  gcint* idxA = idx + ci * m;
  gcint* idxB = idx + cj * m;

  int tid = threadIdx.x;
  float inv = 1.0f / (float)m;
  for (int k = tid; k < m; k += BLOCK) { s_alpha[k] = inv; s_beta[k] = inv; }
  __syncthreads();

  float lr = LR0, prev = 1e30f;
  for (int t = 0; t < MAX_ITE; ++t) {
    compute_diff(X, idxA, idxB, s_alpha, s_beta, s_diff, s_red, m);
    float o = (tid < D) ? s_diff[tid] * s_diff[tid] : 0.0f;
    float obj = blockSum(o, s_red);
    float gap = fabsf(obj - prev);
    prev = obj;

    // B operand (diff) is shared by both gradient passes: build once.
    v16h bmat[8];
    build_bmat(s_diff, bmat);

    // reference applies this iteration's update before the done flag latches
    grad_update(X, idxA, s_alpha, bmat,  2.0f * lr, m);
    project_simplex_lds(s_alpha, s_red, m);
    grad_update(X, idxB, s_beta,  bmat, -2.0f * lr, m);
    project_simplex_lds(s_beta,  s_red, m);

    if (gap < MIN_GAP) break;            // frozen for all remaining iterations
    if (((t + 1) % DECAY_EVERY) == 0) lr *= DECAY;
  }

  // final diff / distance with converged alpha, beta
  compute_diff(X, idxA, idxB, s_alpha, s_beta, s_diff, s_red, m);
  float o = (tid < D) ? s_diff[tid] * s_diff[tid] : 0.0f;
  float obj = blockSum(o, s_red);
  if (tid == 0) dis[p] = obj;
}

__global__ void finalize_kernel(const float* __restrict__ dis,
                                float* __restrict__ out, int P) {
  float s = 0.0f;
  for (int i = threadIdx.x; i < P; i += 32) s += dis[i];
  s = waveRedSum(s);
  if (threadIdx.x == 0) out[0] = -s / (float)P;
}

// ---------------- host entry ----------------

extern "C" void kernel_launch(void* const* d_in, const int* in_sizes, int n_in,
                              void* d_out, int out_size, void* d_ws, size_t ws_size,
                              hipStream_t stream) {
  (void)n_in; (void)out_size; (void)ws_size;
  const float* X = (const float*)d_in[0];
  const int*   y = (const int*)d_in[1];
  int n = in_sizes[1];
  int m = n / NUM_CLASSES;                       // 10000 (multiple of 16)
  int P = NUM_CLASSES * (NUM_CLASSES - 1) / 2;   // 45

  int*   idx = (int*)d_ws;                                        // C*m ints
  float* dis = (float*)((char*)d_ws + (size_t)NUM_CLASSES * m * sizeof(int));

  build_idx_kernel<<<1, 32, 0, stream>>>(y, idx, n, m);

  size_t lds_bytes = (size_t)(2 * m + D + BLOCK) * sizeof(float); // ~83 KB
  vc_pair_kernel<<<P, BLOCK, lds_bytes, stream>>>(X, idx, dis, m);

  finalize_kernel<<<1, 32, 0, stream>>>(dis, (float*)d_out, P);
}
